// CoreDecoder_48696339202073
// MI455X (gfx1250) — matrix operand order repaired and checked by reading, not compiled, not run on hardware
//
#include <hip/hip_runtime.h>
#include <hip/hip_bf16.h>

// ---------------------------------------------------------------------------
// CDNA5 (gfx1250) LPCNet-style decoder.
// All matmuls on v_wmma_f32_16x16x32_bf16 (wave32, 16x16 tiles, K=32).
// GRU recurrence: persistent per-batch-tile kernel, Whh resident in LDS as
// preloaded B fragments; fast rcp-based sigmoid/tanh on the critical path.
// ---------------------------------------------------------------------------

typedef __attribute__((ext_vector_type(16))) __bf16 v16bf;
typedef __attribute__((ext_vector_type(8)))  __bf16 v8bf;
typedef __attribute__((ext_vector_type(8)))  float  v8f;

#define DEVI __device__ __forceinline__

static constexpr int TT   = 1024;   // sequence length
static constexpr int BB   = 128;    // batch
static constexpr int LDX  = 576;    // X activation buffer channel stride
static constexpr int HH   = 64;     // GRU hidden

struct alignas(16) F4 { float v[4]; };
struct alignas(16) U4 { unsigned int v[4]; };
struct alignas(8)  U2 { unsigned int v[2]; };

DEVI unsigned short f2bf(float f) {           // f32 -> bf16 bits, RNE
    unsigned int u = __float_as_uint(f);
    u += 0x7fffu + ((u >> 16) & 1u);
    return (unsigned short)(u >> 16);
}
DEVI unsigned int pack2(float a, float b) {   // two f32 -> packed bf16 dword
    return (unsigned int)f2bf(a) | ((unsigned int)f2bf(b) << 16);
}

DEVI v8bf ld8(const unsigned short* p) {      // 16B-aligned LDS fragment load
    return *reinterpret_cast<const v8bf*>(p);
}
DEVI v16bf mk16(v8bf lo, v8bf hi) {
    union { v16bf v; v8bf h[2]; } u;
    u.h[0] = lo; u.h[1] = hi;
    return u.v;
}
DEVI v8f wmma_bf16(v16bf a, v16bf b, v8f c) {
    return __builtin_amdgcn_wmma_f32_16x16x32_bf16(false, a, false, b,
                                                   (short)0, c, false, false);
}

// fast transcendentals: v_rcp_f32 + v_exp_f32, no IEEE divide expansion
DEVI float frcp(float x)  { return __builtin_amdgcn_rcpf(x); }
DEVI float sigm(float x)  { return frcp(1.f + __expf(-x)); }
DEVI float ftanh(float x) { return 1.f - 2.f * frcp(__expf(2.f * x) + 1.f); }

// ---------------------------------------------------------------------------
// Generic tiled GEMM:  out[M,N] = act( A[M,K] @ W[N,K]^T + bias )
//   M = B*T rows.  Block: 256 thr (8 waves), BM=128, BN=16, BK=32.
//   ASRC: 0 = f32 global (stride K)         [dense1 input z]
//         1 = bf16 X buffer (stride 576)    [Wih projections, output proj]
//         2 = conv tap pair: k<C -> x[t-1 clamped], k>=C -> x[t]  (K = 2C)
//   WSRC: 0 = row-major f32 [N,K]; 1 = conv-interleaved f32 [N,C,2]
//   ACT : 0 = none, 1 = tanh
//   SINK: 0 = bf16 into X buffer at column ocol; 1 = f32 global (stride ldo)
// ---------------------------------------------------------------------------
template<int ASRC, int WSRC, int ACT, int SINK>
__global__ __launch_bounds__(256)
void gemm_kernel(const float* __restrict__ Aimg,
                 const unsigned short* __restrict__ Xin,
                 const float* __restrict__ W,
                 const float* __restrict__ bias,
                 unsigned short* __restrict__ Xout,
                 float* __restrict__ Fout,
                 int K, int ldo, int convC, int ocol)
{
    __shared__ alignas(16) unsigned short As[128][40];  // BM x BK (+pad)
    __shared__ alignas(16) unsigned short Ws[16][40];   // BN x BK (+pad)
    __shared__ alignas(16) float Bs[16];

    const int tid   = threadIdx.x;
    const int wave  = tid >> 5;
    const int lane  = tid & 31;
    const int ntile = blockIdx.x;
    const int mbase = blockIdx.y * 128;

    if (tid < 16) Bs[tid] = bias[ntile * 16 + tid];

    const int row = tid >> 1;            // A-staging mapping
    const int cb  = (tid & 1) * 16;
    const int grow = mbase + row;

    v8f acc = {};
    const int nk = (K + 31) >> 5;

    for (int kc = 0; kc < nk; ++kc) {
        const int k0 = kc * 32;
        __syncthreads();

        // ---- stage A tile (128x32 bf16), vectorized b128 path
        {
            const int kg = k0 + cb;
            if (ASRC == 0) {
                if (kg + 15 < K) {
                    const F4* s = reinterpret_cast<const F4*>(Aimg + (size_t)grow * K + kg);
                    F4 f0 = s[0], f1 = s[1], f2 = s[2], f3 = s[3];
                    U4 u0, u1;
                    u0.v[0] = pack2(f0.v[0], f0.v[1]); u0.v[1] = pack2(f0.v[2], f0.v[3]);
                    u0.v[2] = pack2(f1.v[0], f1.v[1]); u0.v[3] = pack2(f1.v[2], f1.v[3]);
                    u1.v[0] = pack2(f2.v[0], f2.v[1]); u1.v[1] = pack2(f2.v[2], f2.v[3]);
                    u1.v[2] = pack2(f3.v[0], f3.v[1]); u1.v[3] = pack2(f3.v[2], f3.v[3]);
                    *reinterpret_cast<U4*>(&As[row][cb])     = u0;
                    *reinterpret_cast<U4*>(&As[row][cb + 8]) = u1;
                } else {
                    #pragma unroll
                    for (int j = 0; j < 16; ++j) {
                        int k = kg + j;
                        float v = (k < K) ? Aimg[(size_t)grow * K + k] : 0.f;
                        As[row][cb + j] = f2bf(v);
                    }
                }
            } else if (ASRC == 1) {
                // K always a multiple of 32 here -> full group
                const unsigned short* s = Xin + (size_t)grow * LDX + kg;
                if (kg + 32 < K)  // prefetch next K-chunk of this row
                    __builtin_prefetch(s + 32, 0, 3);
                *reinterpret_cast<U4*>(&As[row][cb])     = reinterpret_cast<const U4*>(s)[0];
                *reinterpret_cast<U4*>(&As[row][cb + 8]) = reinterpret_cast<const U4*>(s)[1];
            } else {  // conv tap pair; convC % 16 == 0 -> group in single tap
                const int t  = grow & (TT - 1);
                const size_t cur = (size_t)grow * LDX;
                const size_t prv = (t == 0) ? cur : cur - LDX;
                const unsigned short* s = (kg < convC) ? (Xin + prv + kg)
                                                       : (Xin + cur + (kg - convC));
                *reinterpret_cast<U4*>(&As[row][cb])     = reinterpret_cast<const U4*>(s)[0];
                *reinterpret_cast<U4*>(&As[row][cb + 8]) = reinterpret_cast<const U4*>(s)[1];
            }
        }
        // ---- stage W tile (16x32): f32 -> bf16, 2 elems per thread
        {
            #pragma unroll
            for (int r = 0; r < 2; ++r) {
                int id = tid + r * 256;
                int n  = id >> 5, k = id & 31;
                int gn = ntile * 16 + n, gk = k0 + k;
                float w = 0.f;
                if (gk < K) {
                    if (WSRC == 0) {
                        w = W[(size_t)gn * K + gk];
                    } else {
                        int tap = (gk >= convC) ? 1 : 0;
                        int c   = gk - tap * convC;
                        w = W[((size_t)gn * convC + c) * 2 + tap];
                    }
                }
                Ws[n][k] = f2bf(w);
            }
        }
        __syncthreads();

        // ---- fragments (ISA 16-bit A/B VGPR layouts) + WMMA
        const int arow = wave * 16 + (lane & 15);
        const int akb  = (lane >= 16) ? 8 : 0;
        v16bf a = mk16(ld8(&As[arow][akb]), ld8(&As[arow][16 + akb]));

        const int bn  = lane & 15;
        const int bkb = (lane >= 16) ? 8 : 0;
        v16bf b = mk16(ld8(&Ws[bn][bkb]), ld8(&Ws[bn][16 + bkb]));

        acc = wmma_bf16(a, b, acc);
    }

    // ---- epilogue: C layout = lane(0-15):M=v, lane(16-31):M=8+v, col=lane&15
    const int col   = ntile * 16 + (lane & 15);
    const float bv  = Bs[lane & 15];
    const int rbase = mbase + wave * 16 + ((lane >= 16) ? 8 : 0);
    #pragma unroll
    for (int v = 0; v < 8; ++v) {
        float o = acc[v] + bv;
        if (ACT == 1) o = ftanh(o);
        const int gr = rbase + v;
        if (SINK == 0)
            Xout[(size_t)gr * LDX + ocol + col] = f2bf(o);
        else
            Fout[(size_t)gr * ldo + col] = o;
    }
}

// ---------------------------------------------------------------------------
// GRU recurrence.  grid = 8 blocks (16 batch rows each), block = 128 (4 waves).
// Per step: hg[16,192] = h[16,64] @ Whh^T via 6 WMMAs/wave (3 N-tiles x K=64),
// then fused vectorized gate math; h kept in LDS (f32 + bf16 mirror).
// ---------------------------------------------------------------------------
__global__ __launch_bounds__(128)
void gru_kernel(const float* __restrict__ Whh,   // [192,64]
                const float* __restrict__ bhh,   // [192]
                const float* __restrict__ xw,    // [B,T,192] f32
                const float* __restrict__ gs,    // [B,320] initial states
                int layer,
                unsigned short* __restrict__ Xbuf, int gcol)
{
    __shared__ alignas(16) unsigned short WhhS[192][64];  // bf16, 24 KB
    __shared__ alignas(16) float bhhS[192];
    __shared__ alignas(16) unsigned short hB[16][64];     // h (bf16 mirror)
    __shared__ alignas(16) float hF[16][64];              // h (f32)
    __shared__ alignas(16) float hgS[16][192];            // h @ Whh^T

    const int tid  = threadIdx.x;
    const int wave = tid >> 5;
    const int lane = tid & 31;
    const int b0   = blockIdx.x * 16;

    // Whh -> bf16 LDS, vectorized (rows of 64, groups of 4 never cross rows)
    for (int i = tid * 4; i < 192 * 64; i += 128 * 4) {
        F4 w = *reinterpret_cast<const F4*>(Whh + i);
        U2 u;
        u.v[0] = pack2(w.v[0], w.v[1]);
        u.v[1] = pack2(w.v[2], w.v[3]);
        *reinterpret_cast<U2*>(&WhhS[0][0] + i) = u;
    }
    for (int i = tid; i < 192; i += 128) bhhS[i] = bhh[i];
    for (int i = tid; i < 16 * 64; i += 128) {
        int r = i >> 6, c = i & 63;
        float h0 = gs[(size_t)(b0 + r) * 320 + layer * HH + c];
        hF[r][c] = h0;
        hB[r][c] = f2bf(h0);
    }
    __syncthreads();

    // preload Whh^T B-fragments: wave w owns N-tiles {w, w+4, w+8}
    v16bf bf[3][2];
    #pragma unroll
    for (int j = 0; j < 3; ++j) {
        const int n = (wave + j * 4) * 16 + (lane & 15);
        #pragma unroll
        for (int c = 0; c < 2; ++c) {
            const int kb = c * 32 + ((lane >= 16) ? 8 : 0);
            bf[j][c] = mk16(ld8(&WhhS[n][kb]), ld8(&WhhS[n][kb + 16]));
        }
    }

    const int grow_r = tid >> 3;          // gate-phase mapping: row, 8 cols
    const int gcb    = (tid & 7) * 8;

    for (int t = 0; t < TT; ++t) {
        // A fragments from current h (bf16 in LDS)
        const int ar = lane & 15;
        const int kb = (lane >= 16) ? 8 : 0;
        v16bf a0 = mk16(ld8(&hB[ar][kb]),      ld8(&hB[ar][16 + kb]));
        v16bf a1 = mk16(ld8(&hB[ar][32 + kb]), ld8(&hB[ar][48 + kb]));

        #pragma unroll
        for (int j = 0; j < 3; ++j) {
            v8f acc = {};
            acc = wmma_bf16(a0, bf[j][0], acc);
            acc = wmma_bf16(a1, bf[j][1], acc);
            const int coln = (wave + j * 4) * 16 + (lane & 15);
            const int rb   = (lane >= 16) ? 8 : 0;
            #pragma unroll
            for (int v = 0; v < 8; ++v) hgS[rb + v][coln] = acc[v];
        }
        __syncthreads();

        // fused gates, vectorized 2 x float4 per thread
        const size_t xbase = ((size_t)(b0 + grow_r) * TT + t) * 192;
        const size_t obase = ((size_t)(b0 + grow_r) * TT + t) * LDX + gcol;
        float hv[8];
        #pragma unroll
        for (int g = 0; g < 2; ++g) {
            const int c0 = gcb + g * 4;
            F4 xr = *reinterpret_cast<const F4*>(xw + xbase + c0);
            F4 xz = *reinterpret_cast<const F4*>(xw + xbase + 64 + c0);
            F4 xn = *reinterpret_cast<const F4*>(xw + xbase + 128 + c0);
            F4 hr = *reinterpret_cast<const F4*>(&hgS[grow_r][c0]);
            F4 hz = *reinterpret_cast<const F4*>(&hgS[grow_r][64 + c0]);
            F4 hn = *reinterpret_cast<const F4*>(&hgS[grow_r][128 + c0]);
            F4 br = *reinterpret_cast<const F4*>(&bhhS[c0]);
            F4 bz = *reinterpret_cast<const F4*>(&bhhS[64 + c0]);
            F4 bn = *reinterpret_cast<const F4*>(&bhhS[128 + c0]);
            F4 ho = *reinterpret_cast<const F4*>(&hF[grow_r][c0]);
            F4 hnew;
            #pragma unroll
            for (int q = 0; q < 4; ++q) {
                float r = sigm(xr.v[q] + hr.v[q] + br.v[q]);
                float z = sigm(xz.v[q] + hz.v[q] + bz.v[q]);
                float n = ftanh(xn.v[q] + r * (hn.v[q] + bn.v[q]));
                hnew.v[q] = (1.f - z) * n + z * ho.v[q];
                hv[g * 4 + q] = hnew.v[q];
            }
            *reinterpret_cast<F4*>(&hF[grow_r][c0]) = hnew;
        }
        // pack h -> bf16, single b128 store each to LDS mirror and X buffer
        U4 hu;
        #pragma unroll
        for (int q = 0; q < 4; ++q) hu.v[q] = pack2(hv[2 * q], hv[2 * q + 1]);
        *reinterpret_cast<U4*>(&hB[grow_r][gcb]) = hu;
        *reinterpret_cast<U4*>(Xbuf + obase + gcb) = hu;
        __syncthreads();
    }
}

// ---------------------------------------------------------------------------
// Initial states: gs[b,:] = tanh(W2 @ tanh(W1 @ is[b] + b1) + b2).  Tiny.
// ---------------------------------------------------------------------------
__global__ __launch_bounds__(128)
void init_states(const float* __restrict__ is,   // [128,24]
                 const float* __restrict__ W1, const float* __restrict__ b1,  // [128,24],[128]
                 const float* __restrict__ W2, const float* __restrict__ b2,  // [320,128],[320]
                 float* __restrict__ gs)          // [128,320]
{
    __shared__ float hid[128][128];  // 64 KB: per-batch hidden
    const int b = threadIdx.x;
    for (int j = 0; j < 128; ++j) {
        float s = b1[j];
        for (int k = 0; k < 24; ++k) s += is[b * 24 + k] * W1[j * 24 + k];
        hid[b][j] = ftanh(s);
    }
    for (int j = 0; j < 320; ++j) {
        float s = b2[j];
        for (int k = 0; k < 128; ++k) s += hid[b][k] * W2[j * 128 + k];
        gs[(size_t)b * 320 + j] = ftanh(s);
    }
}

// ---------------------------------------------------------------------------
extern "C" void kernel_launch(void* const* d_in, const int* in_sizes, int n_in,
                              void* d_out, int out_size, void* d_ws, size_t ws_size,
                              hipStream_t stream)
{
    (void)in_sizes; (void)n_in; (void)out_size; (void)ws_size;

    const float* z    = (const float*)d_in[0];   // [128,1024,80]
    const float* ist  = (const float*)d_in[1];   // [128,1,24]
    const float* d1W  = (const float*)d_in[2];
    const float* d1b  = (const float*)d_in[3];
    const float* hiW  = (const float*)d_in[4];
    const float* hib  = (const float*)d_in[5];
    const float* giW  = (const float*)d_in[6];
    const float* gib  = (const float*)d_in[7];
    const float* outW = (const float*)d_in[38];
    const float* outb = (const float*)d_in[39];

    // workspace layout
    const size_t XB_BYTES = (size_t)BB * TT * LDX * sizeof(unsigned short); // ~151 MB
    const size_t XG_BYTES = (size_t)BB * TT * 192 * sizeof(float);          // ~101 MB
    unsigned short* Xbuf = (unsigned short*)d_ws;
    float* xg = (float*)((char*)d_ws + XB_BYTES);
    float* gs = (float*)((char*)d_ws + XB_BYTES + XG_BYTES);

    const int mblocks = (BB * TT) / 128;   // 1024

    // h0's for all 5 GRU layers
    init_states<<<1, 128, 0, stream>>>(ist, hiW, hib, giW, gib, gs);

    // dense1: z(f32,K=80) -> tanh -> X cols [0,96)
    gemm_kernel<0, 0, 1, 0><<<dim3(6, mblocks), 256, 0, stream>>>(
        z, nullptr, d1W, d1b, Xbuf, nullptr, 80, 0, 0, 0);

    for (int i = 0; i < 5; ++i) {
        const int base = 8 + i * 6;
        const float* Wih = (const float*)d_in[base + 0];
        const float* Whh = (const float*)d_in[base + 1];
        const float* bih = (const float*)d_in[base + 2];
        const float* bhh = (const float*)d_in[base + 3];
        const float* cW  = (const float*)d_in[base + 4];
        const float* cb  = (const float*)d_in[base + 5];
        const int Kin = 96 + 96 * i;     // GRU input width
        const int C   = Kin + 64;        // channels after concat(g)

        // xw = X[:, :Kin] @ Wih^T + bih   (f32 out, [B,T,192])
        gemm_kernel<1, 0, 0, 1><<<dim3(12, mblocks), 256, 0, stream>>>(
            nullptr, Xbuf, Wih, bih, nullptr, xg, Kin, 192, 0, 0);

        // GRU scan -> g into X cols [Kin, Kin+64)
        gru_kernel<<<8, 128, 0, stream>>>(Whh, bhh, xg, gs, i, Xbuf, Kin);

        // causal conv k=2 (K=2C) -> tanh -> X cols [C, C+32)
        gemm_kernel<2, 1, 1, 0><<<dim3(2, mblocks), 256, 0, stream>>>(
            nullptr, Xbuf, cW, cb, Xbuf, nullptr, 2 * C, 0, C, C);
    }

    // output projection: X[:,:576] @ outW^T + outb -> d_out f32 [B,T,80]
    gemm_kernel<1, 0, 0, 1><<<dim3(5, mblocks), 256, 0, stream>>>(
        nullptr, Xbuf, outW, outb, nullptr, (float*)d_out, 576, 80, 0, 0);
}
